// CirculatePairConLoss_68152541053485
// MI455X (gfx1250) — matrix operand order repaired and checked by reading, not compiled, not run on hardware
//
#include <hip/hip_runtime.h>
#include <hip/hip_bf16.h>
#include <math.h>

// ---------------------------------------------------------------------------
// CirculatePairConLoss on MI455X (gfx1250, wave32, WMMA + async LDS copy)
//
//   feats = concat(f1,f2) * sqrt(20*log2(e))  as f16   [8192 x 64]
//   per row i:  total_i = sum_j exp2(<F_i,F_j>)        (== exp(dot/T))
//               same_i  = sum_{label_j==label_i} exp2(<F_i,F_j>)  (incl diag)
//   Ng_i   = total_i - same_i
//   loss   = sum_i ( log(Ng_i + pos_i) - posdot_i ) / (2*count[label_i])
// ---------------------------------------------------------------------------

#define N_ROWS   4096
#define TWO_N    8192
#define D_DIM    64
#define N_CLASS  16
#define ROW_TILE 128          // rows per workgroup (8 waves x 16 rows)
#define COL_TILE 64           // columns staged in LDS per iteration
#define COL_SPLITS 8          // grid.y column partitions
#define COLS_PER_SPLIT (TWO_N / COL_SPLITS)   // 1024
#define TILES_PER_SPLIT (COLS_PER_SPLIT / COL_TILE) // 16

// sqrt(20 / ln(2)) : folds 1/T and log2(e) into the f16 data so that
// exp(dot/T) == exp2(<scaled_a, scaled_b>)  -> one v_exp_f32 per element.
#define FEAT_SCALE 5.3715827f
#define INV_T      20.0f

typedef __attribute__((ext_vector_type(16))) _Float16 v16h;
typedef __attribute__((ext_vector_type(8)))  float    v8f;
typedef int v4i __attribute__((vector_size(16)));   // matches builtin param type

union H16 { v16h v; float4 q[2]; };

#define AS_GLOBAL __attribute__((address_space(1)))
#define AS_LDS    __attribute__((address_space(3)))
// address-space casts routed through an integer (always legal in clang);
// low 32 bits of a generic LDS pointer are the LDS byte offset (ISA 10.2).
#define TO_GLOBAL(T, p) ((AS_GLOBAL T*)(unsigned long long)(p))
#define TO_LDS(T, p)    ((AS_LDS T*)(unsigned)(unsigned long long)(p))

#if __has_builtin(__builtin_amdgcn_global_load_async_to_lds_b128) && \
    __has_builtin(__builtin_amdgcn_global_load_async_to_lds_b32)
#define HAVE_ASYNC_LDS 1
#else
#define HAVE_ASYNC_LDS 0
#endif

// ---------------- workspace layout (bytes) ----------------
#define WS_FEATS   0            // f16 [8192][64]  (1,048,576)
#define WS_TOTAL   1048576      // f32 [8192]
#define WS_SAME    1081344      // f32 [8192]
#define WS_POSEXP  1114112      // f32 [4096]
#define WS_POSDOT  1130496      // f32 [4096]

// ---------------------------------------------------------------------------
__global__ void cpcl_init_kernel(float* __restrict__ rs_total,
                                 float* __restrict__ rs_same) {
    int i = blockIdx.x * blockDim.x + threadIdx.x;   // 8192 threads
    rs_total[i] = 0.0f;
    rs_same[i]  = 0.0f;
}

// one block (64 threads) per source row: f32 -> scaled f16 + exact pos dot
__global__ void cpcl_prep_kernel(const float* __restrict__ f1,
                                 const float* __restrict__ f2,
                                 _Float16* __restrict__ feats,
                                 float* __restrict__ pos_exp,
                                 float* __restrict__ pos_dot) {
    int r = blockIdx.x;          // 0..4095
    int d = threadIdx.x;         // 0..63
    float a = f1[r * D_DIM + d];
    float b = f2[r * D_DIM + d];
    feats[r * D_DIM + d]            = (_Float16)(a * FEAT_SCALE);
    feats[(N_ROWS + r) * D_DIM + d] = (_Float16)(b * FEAT_SCALE);

    __shared__ float red[64];
    red[d] = a * b;
    __syncthreads();
    for (int s = 32; s > 0; s >>= 1) {
        if (d < s) red[d] += red[d + s];
        __syncthreads();
    }
    if (d == 0) {
        float pd = red[0] * INV_T;
        pos_dot[r] = pd;
        pos_exp[r] = expf(pd);
    }
}

// ---------------------------------------------------------------------------
// Main kernel: masked exp row-sums of feats @ feats^T via v_wmma_f32_16x16x32_f16.
// Column tiles stream memory -> LDS with GLOBAL_LOAD_ASYNC_TO_LDS (ASYNCcnt),
// double-buffered so the copy of tile t+1 overlaps the WMMA+exp of tile t.
// block = 256 threads (8 waves), each wave owns 16 rows; grid = (64, 8).
// ---------------------------------------------------------------------------
__device__ __forceinline__
void cpcl_stage_tile(const _Float16* __restrict__ feats,
                     const int* __restrict__ label,
                     _Float16* __restrict__ ldsF,
                     int* __restrict__ ldsLab,
                     int colBase, int tid) {
#if HAVE_ASYNC_LDS
    // 64 cols x 64 halves = 512 x 16B chunks; 256 threads -> 2 b128 asyncs each
#pragma unroll
    for (int c = tid; c < COL_TILE * 8; c += 256) {
        int col = c >> 3;
        int k8  = (c & 7) * 8;
        __builtin_amdgcn_global_load_async_to_lds_b128(
            TO_GLOBAL(v4i, &feats[(colBase + col) * D_DIM + k8]),
            TO_LDS(v4i, &ldsF[col * D_DIM + k8]),
            0, 0);
    }
    if (tid < COL_TILE) {
        __builtin_amdgcn_global_load_async_to_lds_b32(
            TO_GLOBAL(int, &label[(colBase + tid) & (N_ROWS - 1)]),
            TO_LDS(int, &ldsLab[tid]),
            0, 0);
    }
#else
#pragma unroll
    for (int c = tid; c < COL_TILE * 8; c += 256) {
        int col = c >> 3;
        int k8  = (c & 7) * 8;
        *(float4*)&ldsF[col * D_DIM + k8] =
            *(const float4*)&feats[(colBase + col) * D_DIM + k8];
    }
    if (tid < COL_TILE)
        ldsLab[tid] = label[(colBase + tid) & (N_ROWS - 1)];
#endif
}

__device__ __forceinline__ void cpcl_wait_stage() {
#if HAVE_ASYNC_LDS
    asm volatile("s_wait_asynccnt 0x0" ::: "memory");
#endif
}

__global__ void __launch_bounds__(256)
cpcl_gemm_kernel(const _Float16* __restrict__ feats,
                 const int* __restrict__ label,
                 float* __restrict__ rs_total,
                 float* __restrict__ rs_same) {
    __shared__ __align__(16) _Float16 ldsF[2][COL_TILE * D_DIM];  // 2 x 8 KB
    __shared__ int ldsLab[2][COL_TILE];

    const int tid     = threadIdx.x;
    const int lane    = tid & 31;
    const int wave    = tid >> 5;
    const int halfSel = (lane >> 4) & 1;     // 0: lanes 0-15, 1: lanes 16-31
    const int lane16  = lane & 15;

    const int rowWave  = blockIdx.x * ROW_TILE + wave * 16;   // 16 rows per wave
    const int colBase0 = blockIdx.y * COLS_PER_SPLIT;

    // kick off the async copy of tile 0 before doing anything else
    cpcl_stage_tile(feats, label, ldsF[0], ldsLab[0], colBase0, tid);

    // ---- A fragments (this wave's 16 rows, K=0..63), resident for whole kernel.
    // 16-bit A 16x32 layout: lane m (0-15) holds K chunks {0..7,16..23},
    // lane m+16 holds {8..15,24..31}; a_hi is the same pattern shifted by 32.
    const int rowA = rowWave + lane16;
    const _Float16* rp = feats + rowA * D_DIM + 8 * halfSel;
    H16 aLo, aHi;
    aLo.q[0] = *(const float4*)(rp + 0);
    aLo.q[1] = *(const float4*)(rp + 16);
    aHi.q[0] = *(const float4*)(rp + 32);
    aHi.q[1] = *(const float4*)(rp + 48);

    // ---- labels of the 8 rows this lane accumulates (C layout: M = r + 8*halfSel)
    int rowLab[8];
#pragma unroll
    for (int r = 0; r < 8; ++r)
        rowLab[r] = label[(rowWave + 8 * halfSel + r) & (N_ROWS - 1)];

    float accT[8], accS[8];
#pragma unroll
    for (int r = 0; r < 8; ++r) { accT[r] = 0.0f; accS[r] = 0.0f; }

    for (int t = 0; t < TILES_PER_SPLIT; ++t) {
        const int cur = t & 1;

        // my async chunks for tile t are done; barrier publishes everyone's.
        // The same barrier also orders all waves' compute of tile t-1 before
        // anyone overwrites that buffer below.
        cpcl_wait_stage();
        __syncthreads();

        if (t + 1 < TILES_PER_SPLIT)
            cpcl_stage_tile(feats, label, ldsF[cur ^ 1], ldsLab[cur ^ 1],
                            colBase0 + (t + 1) * COL_TILE, tid);

        // ---- compute on tile t: 4 sub-tiles of 16 columns each
#pragma unroll
        for (int sub = 0; sub < 4; ++sub) {
            // 16-bit B 32x16 layout: lane n (0-15) holds K {0..7,16..23} of column n,
            // lane n+16 holds {8..15,24..31}; b_hi same pattern shifted by 32.
            const _Float16* bp =
                &ldsF[cur][(sub * 16 + lane16) * D_DIM + 8 * halfSel];
            H16 bLo, bHi;
            bLo.q[0] = *(const float4*)(bp + 0);
            bLo.q[1] = *(const float4*)(bp + 16);
            bHi.q[0] = *(const float4*)(bp + 32);
            bHi.q[1] = *(const float4*)(bp + 48);

            v8f c = {};
            c = __builtin_amdgcn_wmma_f32_16x16x32_f16(false, aLo.v, false, bLo.v,
                                                       (short)0, c, false, false);
            c = __builtin_amdgcn_wmma_f32_16x16x32_f16(false, aHi.v, false, bHi.v,
                                                       (short)0, c, false, false);

            int colLab = ldsLab[cur][sub * 16 + lane16];
#pragma unroll
            for (int r = 0; r < 8; ++r) {
                float e = __builtin_amdgcn_exp2f(c[r]);   // exp(dot/T), scale folded
                accT[r] += e;
                accS[r] += (rowLab[r] == colLab) ? e : 0.0f;
            }
        }
    }

    // reduce each row across the 16 lanes that hold its columns, then commit
#pragma unroll
    for (int r = 0; r < 8; ++r) {
        float tv = accT[r], sv = accS[r];
#pragma unroll
        for (int m = 1; m < 16; m <<= 1) {
            tv += __shfl_xor(tv, m, 32);
            sv += __shfl_xor(sv, m, 32);
        }
        if (lane16 == 0) {
            int row = rowWave + 8 * halfSel + r;
            atomicAdd(&rs_total[row], tv);
            atomicAdd(&rs_same[row], sv);
        }
    }
}

// ---------------------------------------------------------------------------
// Final: one block folds 8192 row terms into the scalar loss.
// ---------------------------------------------------------------------------
__global__ void cpcl_loss_kernel(const float* __restrict__ rs_total,
                                 const float* __restrict__ rs_same,
                                 const float* __restrict__ pos_exp,
                                 const float* __restrict__ pos_dot,
                                 const int* __restrict__ label,
                                 float* __restrict__ out) {
    __shared__ int   cnt[N_CLASS];
    __shared__ float red[256];
    int t = threadIdx.x;
    if (t < N_CLASS) cnt[t] = 0;
    __syncthreads();
    for (int i = t; i < N_ROWS; i += 256)
        atomicAdd(&cnt[label[i]], 1);
    __syncthreads();

    float local = 0.0f;
    for (int i = t; i < TWO_N; i += 256) {
        int   src  = i & (N_ROWS - 1);
        float Ng   = rs_total[i] - rs_same[i];        // excludes same-label (and diag)
        float p    = pos_exp[src];
        float term = logf(Ng + p) - pos_dot[src];     // -log(p/(Ng+p))
        int   gs   = 2 * cnt[label[src]];
        local += term / (float)gs;
    }
    red[t] = local;
    __syncthreads();
    for (int s = 128; s > 0; s >>= 1) {
        if (t < s) red[t] += red[t + s];
        __syncthreads();
    }
    if (t == 0) out[0] = red[0];
}

// ---------------------------------------------------------------------------
extern "C" void kernel_launch(void* const* d_in, const int* in_sizes, int n_in,
                              void* d_out, int out_size, void* d_ws, size_t ws_size,
                              hipStream_t stream) {
    const float* f1    = (const float*)d_in[0];
    const float* f2    = (const float*)d_in[1];
    const int*   label = (const int*)d_in[2];

    char* ws = (char*)d_ws;
    _Float16* feats   = (_Float16*)(ws + WS_FEATS);
    float*    rsTot   = (float*)(ws + WS_TOTAL);
    float*    rsSame  = (float*)(ws + WS_SAME);
    float*    posExp  = (float*)(ws + WS_POSEXP);
    float*    posDot  = (float*)(ws + WS_POSDOT);
    float*    out     = (float*)d_out;

    cpcl_init_kernel<<<TWO_N / 256, 256, 0, stream>>>(rsTot, rsSame);
    cpcl_prep_kernel<<<N_ROWS, 64, 0, stream>>>(f1, f2, feats, posExp, posDot);
    dim3 grid(TWO_N / ROW_TILE, COL_SPLITS);
    cpcl_gemm_kernel<<<grid, 256, 0, stream>>>(feats, label, rsTot, rsSame);
    cpcl_loss_kernel<<<1, 256, 0, stream>>>(rsTot, rsSame, posExp, posDot, label, out);
}
